// ModulatedConv2d_18056042512936
// MI455X (gfx1250) — compile-verified, matrix operand order checked by reading, not run on hardware
//
#include <hip/hip_runtime.h>
#include <hip/hip_bf16.h>

#define IN_CH 512
#define OUT_CH 512
#define STYLE_DIM 512
#define BATCH 16
#define HWDIM 64
#define EPS_F 1e-8f

// 1/sqrt(512), 1/sqrt(512*9), (1/sqrt(512*9))^2 = 1/4608
#define MOD_SCALE 0.0441941738241592f
#define CONV_SCALE 0.0147313913167829f
#define CONV_SCALE2 (1.0f / 4608.0f)

typedef __attribute__((ext_vector_type(16))) __bf16 v16bf;
typedef __attribute__((ext_vector_type(8))) float v8f;

struct alignas(16) U128 { unsigned int v[4]; };
struct U256 { U128 lo, hi; };

typedef __attribute__((address_space(3))) unsigned short lds_ushort;

__device__ __forceinline__ unsigned lds_off_u32(const unsigned short* p) {
  return (unsigned)(uintptr_t)(lds_ushort*)(unsigned short*)p;
}

__device__ __forceinline__ unsigned short f2bf(float x) {
  unsigned int u = __builtin_bit_cast(unsigned int, x);
  u += 0x7FFFu + ((u >> 16) & 1u);  // round-to-nearest-even
  return (unsigned short)(u >> 16);
}

// ---------------- s[b,i] = style[b,:] @ (modW * MOD_SCALE) + bias[i] ----------------
__global__ __launch_bounds__(512) void modulate_kernel(const float* __restrict__ style,
                                                       const float* __restrict__ modw,
                                                       const float* __restrict__ modb,
                                                       float* __restrict__ s) {
  const int b = blockIdx.x;
  const int i = threadIdx.x;
  const float* st = style + b * STYLE_DIM;
  float acc = 0.0f;
  for (int j = 0; j < STYLE_DIM; ++j) acc += st[j] * modw[j * IN_CH + i];
  s[b * IN_CH + i] = acc * MOD_SCALE + modb[i];
}

// ---------------- wsq[o,i] = sum_tap weight[o,i,tap]^2 ----------------
__global__ __launch_bounds__(256) void wsq_kernel(const float* __restrict__ w,
                                                  float* __restrict__ wsq) {
  const int idx = blockIdx.x * 256 + threadIdx.x;
  if (idx < OUT_CH * IN_CH) {
    const float* p = w + (long)idx * 9;
    float a = 0.0f;
#pragma unroll
    for (int t = 0; t < 9; ++t) a += p[t] * p[t];
    wsq[idx] = a;
  }
}

// ---------------- demod[b,o] = rsqrt(cs^2 * sum_i wsq[o,i]*s[b,i]^2 + eps) ----------------
__global__ __launch_bounds__(256) void demod_kernel(const float* __restrict__ wsq,
                                                    const float* __restrict__ s,
                                                    float* __restrict__ demod) {
  const int o = blockIdx.x * 256 + threadIdx.x;
  const int b = blockIdx.y;
  const float* wr = wsq + (long)o * IN_CH;
  const float* sr = s + b * IN_CH;
  float acc = 0.0f;
  for (int i = 0; i < IN_CH; ++i) {
    float si = sr[i];
    acc += wr[i] * si * si;
  }
  demod[b * OUT_CH + o] = rsqrtf(CONV_SCALE2 * acc + EPS_F);
}

// ---------------- bf16 weight repack: wbf[tap][o][i] ----------------
__global__ __launch_bounds__(256) void repack_kernel(const float* __restrict__ w,
                                                     unsigned short* __restrict__ wbf) {
  const int idx = blockIdx.x * 256 + threadIdx.x;
  if (idx < 9 * OUT_CH * IN_CH) {
    const int t = idx / (OUT_CH * IN_CH);
    const int rem = idx % (OUT_CH * IN_CH);  // o*IN_CH + i
    wbf[idx] = f2bf(w[(long)rem * 9 + t]);
  }
}

// ---------------- main implicit-GEMM conv with bf16 WMMA ----------------
// grid: (OUT_CH/128, HWDIM/2, BATCH); block: 256 threads (8 waves)
// Each block: 128 output channels x 128 pixels (two output rows).
__global__ __launch_bounds__(256) void conv_wmma_kernel(const float* __restrict__ x,
                                                        const unsigned short* __restrict__ wbf,
                                                        const float* __restrict__ s,
                                                        const float* __restrict__ demod,
                                                        float* __restrict__ out) {
  __shared__ unsigned short Asm[128 * 40];  // 128 Cout rows x 32 bf16, row stride 40 (80B)
  __shared__ unsigned short Bsm[128 * 40];  // 128 pixels x 32 bf16
  __shared__ float Ssm[IN_CH];

  const int tid = threadIdx.x;
  const int ctile = blockIdx.x;      // 0..3
  const int h0 = blockIdx.y * 2;     // first of two output rows
  const int b = blockIdx.z;

  Ssm[tid] = s[b * IN_CH + tid];
  Ssm[tid + 256] = s[b * IN_CH + tid + 256];

  const int lane = tid & 31;
  const int wave = tid >> 5;
  const int wm = wave >> 1;           // 0..3 -> 32-row Cout band
  const int wn = wave & 1;            // 0..1 -> 64-pixel band
  const int koff = (lane >> 4) << 3;  // 8-ushort offset for upper half-wave
  const int mrow = lane & 15;

  // A-tile async-copy chunk assignment: 512 chunks of 16B (128 rows x 4), 2 per thread
  const int ck0 = tid * 2;

  // B-tile staging assignment: 2 threads per pixel, 16 channels each
  const int bn = tid & 127;       // pixel 0..127
  const int bcg = (tid >> 7) << 4;  // channel sub-base 0 or 16
  const int brow = bn >> 6;       // which of the two output rows
  const int bcol = bn & 63;

  v8f zero = {0.f, 0.f, 0.f, 0.f, 0.f, 0.f, 0.f, 0.f};
  v8f acc[2][4];
#pragma unroll
  for (int i = 0; i < 2; ++i)
#pragma unroll
    for (int j = 0; j < 4; ++j) acc[i][j] = zero;

  for (int tap = 0; tap < 9; ++tap) {
    const int dh = tap / 3, dw = tap % 3;
    const int ys = h0 + brow + dh - 1;
    const int xs = bcol + dw - 1;
    const bool ok = (ys >= 0) && (ys < HWDIM) && (xs >= 0) && (xs < HWDIM);
    const unsigned short* wtap = wbf + ((long)(tap * OUT_CH + ctile * 128)) * IN_CH;

    for (int cb = 0; cb < IN_CH / 32; ++cb) {
      const int c0 = cb * 32;
      __syncthreads();  // protect previous tile reads (and Ssm on first iter)

      // ---- stage A: 128 x 32 bf16 weights via async global->LDS copy ----
      {
        const unsigned short* saddr = wtap + c0;  // wave-uniform 64-bit base
#pragma unroll
        for (int j = 0; j < 2; ++j) {
          const int kk = ck0 + j;
          const int r = kk >> 2;       // Cout row in tile
          const int p = kk & 3;        // 16B part within the 64B row
          const unsigned lds_addr = lds_off_u32(&Asm[r * 40 + p * 8]);
          const unsigned goff = (unsigned)(r * IN_CH + p * 8) * 2u;  // byte offset
          asm volatile("global_load_async_to_lds_b128 %0, %1, %2"
                       :
                       : "v"(lds_addr), "v"(goff), "s"(saddr)
                       : "memory");
        }
      }
      // ---- stage B: 128 pixels x 32 channels of style-scaled input, fp32 -> packed bf16 ----
      {
        unsigned int* bdst = (unsigned int*)&Bsm[bn * 40 + bcg];
#pragma unroll
        for (int j = 0; j < 8; ++j) {
          const int c = bcg + 2 * j;
          float v0 = 0.0f, v1 = 0.0f;
          if (ok) {
            const float* xp =
                x + (((long)b * IN_CH + c0 + c) * HWDIM + ys) * HWDIM + xs;
            v0 = xp[0] * Ssm[c0 + c];
            v1 = xp[HWDIM * HWDIM] * Ssm[c0 + c + 1];
          }
          bdst[j] = (unsigned)f2bf(v0) | ((unsigned)f2bf(v1) << 16);
        }
      }
      asm volatile("s_wait_asynccnt 0x0" ::: "memory");
      __syncthreads();

      // ---- fragments + WMMA ----
      v16bf afr[2], bfr[4];
#pragma unroll
      for (int sm = 0; sm < 2; ++sm) {
        const unsigned short* ap = &Asm[(wm * 32 + sm * 16 + mrow) * 40 + koff];
        U256 t;
        t.lo = *(const U128*)ap;
        t.hi = *(const U128*)(ap + 16);
        afr[sm] = __builtin_bit_cast(v16bf, t);
      }
#pragma unroll
      for (int sn = 0; sn < 4; ++sn) {
        const unsigned short* bp = &Bsm[(wn * 64 + sn * 16 + mrow) * 40 + koff];
        U256 t;
        t.lo = *(const U128*)bp;
        t.hi = *(const U128*)(bp + 16);
        bfr[sn] = __builtin_bit_cast(v16bf, t);
      }
#pragma unroll
      for (int sm = 0; sm < 2; ++sm)
#pragma unroll
        for (int sn = 0; sn < 4; ++sn)
          acc[sm][sn] = __builtin_amdgcn_wmma_f32_16x16x32_bf16(
              false, afr[sm], false, bfr[sn], (short)0, acc[sm][sn], false, false);
    }
  }

  // ---- epilogue: y *= conv_scale * demod[b,o] ----
#pragma unroll
  for (int sm = 0; sm < 2; ++sm) {
#pragma unroll
    for (int sn = 0; sn < 4; ++sn) {
      const int p = wn * 64 + sn * 16 + (lane & 15);
      const int hh = h0 + (p >> 6);
      const int w = p & 63;
#pragma unroll
      for (int r = 0; r < 8; ++r) {
        const int orow = ctile * 128 + wm * 32 + sm * 16 + r + ((lane >> 4) << 3);
        const float mult = CONV_SCALE * demod[b * OUT_CH + orow];
        out[(((long)b * OUT_CH + orow) * HWDIM + hh) * HWDIM + w] = acc[sm][sn][r] * mult;
      }
    }
  }
}

extern "C" void kernel_launch(void* const* d_in, const int* in_sizes, int n_in,
                              void* d_out, int out_size, void* d_ws, size_t ws_size,
                              hipStream_t stream) {
  const float* x = (const float*)d_in[0];       // [16,512,64,64]
  const float* style = (const float*)d_in[1];   // [16,512]
  const float* weight = (const float*)d_in[2];  // [1,512,512,3,3]
  const float* modw = (const float*)d_in[3];    // [512,512]
  const float* modb = (const float*)d_in[4];    // [512]
  float* out = (float*)d_out;

  char* ws = (char*)d_ws;
  float* s_mod = (float*)ws;                                      // 16*512 f32 (32 KB)
  float* demod = (float*)(ws + 32768);                            // 16*512 f32 (32 KB)
  float* wsq = (float*)(ws + 65536);                              // 512*512 f32 (1 MB)
  unsigned short* wbf = (unsigned short*)(ws + 65536 + 1048576);  // 9*512*512 bf16 (4.5 MB)

  modulate_kernel<<<dim3(BATCH), dim3(IN_CH), 0, stream>>>(style, modw, modb, s_mod);
  wsq_kernel<<<dim3((OUT_CH * IN_CH + 255) / 256), dim3(256), 0, stream>>>(weight, wsq);
  demod_kernel<<<dim3(OUT_CH / 256, BATCH), dim3(256), 0, stream>>>(wsq, s_mod, demod);
  repack_kernel<<<dim3((9 * OUT_CH * IN_CH + 255) / 256), dim3(256), 0, stream>>>(weight, wbf);
  conv_wmma_kernel<<<dim3(OUT_CH / 128, HWDIM / 2, BATCH), dim3(256), 0, stream>>>(
      x, wbf, s_mod, demod, out);
}
